// SparseLinearOld_21466246545932
// MI455X (gfx1250) — matrix operand order repaired and checked by reading, not compiled, not run on hardware
//
#include <hip/hip_runtime.h>
#include <hip/hip_bf16.h>

// ---------------------------------------------------------------------------
// y = X @ (W * mask)^T + b  via bf16x3 split-precision WMMA on gfx1250.
// Pass 1: convert X -> (Xhi,Xlo) bf16 planes; W*mask -> (Whi,Wlo) planes (ws).
// Pass 2: 256x256-tile GEMM, 16 waves/block, 3 bf16 WMMAs per 16x16x32 step.
//         Tiles staged with gfx1250 async global->LDS copies (ASYNCcnt) and
//         double-buffered LDS when the builtins exist; register-staged
//         software pipeline otherwise.
// Falls back to a fused (convert-in-loop) kernel if workspace is too small.
// ---------------------------------------------------------------------------

typedef __attribute__((ext_vector_type(16))) __bf16          v16bf;
typedef __attribute__((ext_vector_type(8)))  float           v8f;
typedef __attribute__((ext_vector_type(8)))  unsigned short  v8us;
typedef __attribute__((ext_vector_type(16))) unsigned short  v16us;
typedef __attribute__((ext_vector_type(4)))  int             v4i;

#define IN_F  4096
#define OUT_F 4096
#define TOK   8192

#if defined(__HIP_DEVICE_COMPILE__) && \
    __has_builtin(__builtin_amdgcn_global_load_async_to_lds_b128)
#define HAVE_ASYNC 1
#else
#define HAVE_ASYNC 0
#endif

#if defined(__HIP_DEVICE_COMPILE__) && \
    __has_builtin(__builtin_amdgcn_s_wait_asynccnt)
#define WAIT_ASYNC(n) __builtin_amdgcn_s_wait_asynccnt(n)
#elif defined(__HIP_DEVICE_COMPILE__)
#define WAIT_ASYNC(n) asm volatile("s_wait_asynccnt %0" ::"i"(n) : "memory")
#else
#define WAIT_ASYNC(n)
#endif

#define GLOBAL_AS __attribute__((address_space(1)))
#define LDS_AS    __attribute__((address_space(3)))

__device__ __forceinline__ unsigned bf16_rne(float f) {
    unsigned u = __float_as_uint(f);
    return (u + 0x7FFFu + ((u >> 16) & 1u)) >> 16;
}

__device__ __forceinline__ void split_pack(float f0, float f1,
                                           unsigned& hp, unsigned& lp) {
    unsigned h0 = bf16_rne(f0), h1 = bf16_rne(f1);
    float r0 = f0 - __uint_as_float(h0 << 16);
    float r1 = f1 - __uint_as_float(h1 << 16);
    unsigned l0 = bf16_rne(r0), l1 = bf16_rne(r1);
    hp = h0 | (h1 << 16);
    lp = l0 | (l1 << 16);
}

__device__ __forceinline__ v16bf ld_frag(const unsigned short* p, int o0, int o1) {
    v8us a = *(const v8us*)(p + o0);
    v8us b = *(const v8us*)(p + o1);
    v16us c = __builtin_shufflevector(a, b, 0, 1, 2, 3, 4, 5, 6, 7,
                                            8, 9, 10, 11, 12, 13, 14, 15);
    return __builtin_bit_cast(v16bf, c);
}

// ========================= pass 1: conversions =============================

__global__ __launch_bounds__(256)
void prep_x(const float* __restrict__ X,
            unsigned short* __restrict__ Xh, unsigned short* __restrict__ Xl) {
    size_t idx = ((size_t)blockIdx.x * 256 + threadIdx.x) * 8;
    float xr[8];
    *(float4*)&xr[0] = ((const float4*)(X + idx))[0];
    *(float4*)&xr[4] = ((const float4*)(X + idx))[1];
    unsigned hp[4], lp[4];
#pragma unroll
    for (int u = 0; u < 4; ++u) split_pack(xr[2 * u], xr[2 * u + 1], hp[u], lp[u]);
    uint4 h4; h4.x = hp[0]; h4.y = hp[1]; h4.z = hp[2]; h4.w = hp[3];
    uint4 l4; l4.x = lp[0]; l4.y = lp[1]; l4.z = lp[2]; l4.w = lp[3];
    *(uint4*)(Xh + idx) = h4;
    *(uint4*)(Xl + idx) = l4;
}

__global__ __launch_bounds__(256)
void prep_w(const float* __restrict__ W, const unsigned char* __restrict__ Mk,
            unsigned short* __restrict__ Wh, unsigned short* __restrict__ Wl) {
    size_t idx = ((size_t)blockIdx.x * 256 + threadIdx.x) * 8;
    float wr[8];
    *(float4*)&wr[0] = ((const float4*)(W + idx))[0];
    *(float4*)&wr[4] = ((const float4*)(W + idx))[1];
    uint2 mv = *(const uint2*)(Mk + idx);
    unsigned mw[2] = {mv.x, mv.y};
#pragma unroll
    for (int u = 0; u < 8; ++u)
        if (((mw[u >> 2] >> ((u & 3) * 8)) & 0xFFu) == 0u) wr[u] = 0.0f;
    unsigned hp[4], lp[4];
#pragma unroll
    for (int u = 0; u < 4; ++u) split_pack(wr[2 * u], wr[2 * u + 1], hp[u], lp[u]);
    uint4 h4; h4.x = hp[0]; h4.y = hp[1]; h4.z = hp[2]; h4.w = hp[3];
    uint4 l4; l4.x = lp[0]; l4.y = lp[1]; l4.z = lp[2]; l4.w = lp[3];
    *(uint4*)(Wh + idx) = h4;
    *(uint4*)(Wl + idx) = l4;
}

// ========================= pass 2: GEMM ====================================
// 256x256 tile, BK=32, 512 threads = 16 waves (4x4). Wave owns 64x64.

#define BM 256
#define BN 256
#define BK 32
#define LDK 40   // padded LDS row stride (ushorts): 80 B/row, spreads banks
#define NKSTEP (IN_F / BK)

__global__ __launch_bounds__(512)
void gemm_bf16x3(const unsigned short* __restrict__ Xh,
                 const unsigned short* __restrict__ Xl,
                 const unsigned short* __restrict__ Wh,
                 const unsigned short* __restrict__ Wl,
                 const float* __restrict__ Bv,
                 float* __restrict__ Y) {
#if HAVE_ASYNC
    __shared__ unsigned short Ahi[2][BM][LDK];
    __shared__ unsigned short Alo[2][BM][LDK];
    __shared__ unsigned short Bhi[2][BN][LDK];
    __shared__ unsigned short Blo[2][BN][LDK];
#else
    __shared__ unsigned short Ahi[1][BM][LDK];
    __shared__ unsigned short Alo[1][BM][LDK];
    __shared__ unsigned short Bhi[1][BN][LDK];
    __shared__ unsigned short Blo[1][BN][LDK];
#endif

    const int tid = threadIdx.x;
    const int m0  = blockIdx.y * BM;
    const int n0  = blockIdx.x * BN;

    const int lane = tid & 31;
    const int wv   = tid >> 5;
    const int wm   = wv >> 2;       // 0..3
    const int wn   = wv & 3;        // 0..3
    const int l16  = lane & 15;
    const int hh   = (lane >> 4) & 1;
    const int aoff = hh * 16;       // A frag: K {0..15} or {16..31}
    const int boff = hh * 16;       // B frag: K {0..15} or {16..31}

    v8f acc[4][4];
#pragma unroll
    for (int i = 0; i < 4; ++i)
#pragma unroll
        for (int j = 0; j < 4; ++j)
#pragma unroll
            for (int v = 0; v < 8; ++v) acc[i][j][v] = 0.0f;

    // copy mapping: 8 x b128 chunks per thread per K-step
    // ch -> (plane = ch>>1, it = ch&1); idx = it*512+tid; row = idx>>2, c=(idx&3)*8

#if HAVE_ASYNC
    // ---- async double-buffered pipeline --------------------------------
    auto issue_tile = [&](int kt, int buf) {
        const int k0 = kt * BK;
#pragma unroll
        for (int ch = 0; ch < 8; ++ch) {
            const int pl  = ch >> 1;
            const int idx = (ch & 1) * 512 + tid;
            const int r   = idx >> 2;
            const int c   = (idx & 3) * 8;
            const unsigned short* src =
                (pl == 0 ? Xh : pl == 1 ? Xl : pl == 2 ? Wh : Wl) +
                (size_t)((pl < 2 ? m0 : n0) + r) * IN_F + k0 + c;
            unsigned short* dst =
                (pl == 0 ? &Ahi[buf][r][c] : pl == 1 ? &Alo[buf][r][c]
               : pl == 2 ? &Bhi[buf][r][c] : &Blo[buf][r][c]);
            __builtin_amdgcn_global_load_async_to_lds_b128(
                (GLOBAL_AS v4i*)src, (LDS_AS v4i*)dst, 0, 0);
        }
    };

    issue_tile(0, 0);
    for (int kt = 0; kt < NKSTEP; ++kt) {
        const int cur = kt & 1;
        if (kt + 1 < NKSTEP) {
            issue_tile(kt + 1, cur ^ 1);   // overlap with this tile's WMMAs
            WAIT_ASYNC(8);                 // in-order: tile kt fully in LDS
        } else {
            WAIT_ASYNC(0);
        }
        __syncthreads();

        v16bf fbh[4], fbl[4];
#pragma unroll
        for (int j = 0; j < 4; ++j) {
            const int nc = wn * 64 + j * 16 + l16;
            fbh[j] = ld_frag(&Bhi[cur][nc][0], boff, boff + 8);
            fbl[j] = ld_frag(&Blo[cur][nc][0], boff, boff + 8);
        }
#pragma unroll
        for (int i = 0; i < 4; ++i) {
            const int mr = wm * 64 + i * 16 + l16;
            v16bf fah = ld_frag(&Ahi[cur][mr][0], aoff, aoff + 8);
            v16bf fal = ld_frag(&Alo[cur][mr][0], aoff, aoff + 8);
#pragma unroll
            for (int j = 0; j < 4; ++j) {
                acc[i][j] = __builtin_amdgcn_wmma_f32_16x16x32_bf16(
                    false, fah, false, fbh[j], (short)0, acc[i][j], false, false);
                acc[i][j] = __builtin_amdgcn_wmma_f32_16x16x32_bf16(
                    false, fah, false, fbl[j], (short)0, acc[i][j], false, false);
                acc[i][j] = __builtin_amdgcn_wmma_f32_16x16x32_bf16(
                    false, fal, false, fbh[j], (short)0, acc[i][j], false, false);
            }
        }
        __syncthreads();   // buf[cur] free for async overwrite next iter
    }
#else
    // ---- register-staged software pipeline (fallback) ------------------
    uint4 rg[8];
#pragma unroll
    for (int ch = 0; ch < 8; ++ch) {
        const int pl  = ch >> 1;
        const int idx = (ch & 1) * 512 + tid;
        const int r   = idx >> 2;
        const int c   = (idx & 3) * 8;
        const unsigned short* src =
            (pl == 0 ? Xh : pl == 1 ? Xl : pl == 2 ? Wh : Wl) +
            (size_t)((pl < 2 ? m0 : n0) + r) * IN_F + c;
        rg[ch] = *(const uint4*)src;
    }

    for (int kt = 0; kt < NKSTEP; ++kt) {
#pragma unroll
        for (int ch = 0; ch < 8; ++ch) {
            const int pl  = ch >> 1;
            const int idx = (ch & 1) * 512 + tid;
            const int r   = idx >> 2;
            const int c   = (idx & 3) * 8;
            unsigned short* dst =
                (pl == 0 ? &Ahi[0][r][c] : pl == 1 ? &Alo[0][r][c]
               : pl == 2 ? &Bhi[0][r][c] : &Blo[0][r][c]);
            *(uint4*)dst = rg[ch];
        }
        __syncthreads();

        if (kt + 1 < NKSTEP) {
            const int k0n = (kt + 1) * BK;
#pragma unroll
            for (int ch = 0; ch < 8; ++ch) {
                const int pl  = ch >> 1;
                const int idx = (ch & 1) * 512 + tid;
                const int r   = idx >> 2;
                const int c   = (idx & 3) * 8;
                const unsigned short* src =
                    (pl == 0 ? Xh : pl == 1 ? Xl : pl == 2 ? Wh : Wl) +
                    (size_t)((pl < 2 ? m0 : n0) + r) * IN_F + k0n + c;
                rg[ch] = *(const uint4*)src;
            }
        }

        v16bf fbh[4], fbl[4];
#pragma unroll
        for (int j = 0; j < 4; ++j) {
            const int nc = wn * 64 + j * 16 + l16;
            fbh[j] = ld_frag(&Bhi[0][nc][0], boff, boff + 8);
            fbl[j] = ld_frag(&Blo[0][nc][0], boff, boff + 8);
        }
#pragma unroll
        for (int i = 0; i < 4; ++i) {
            const int mr = wm * 64 + i * 16 + l16;
            v16bf fah = ld_frag(&Ahi[0][mr][0], aoff, aoff + 8);
            v16bf fal = ld_frag(&Alo[0][mr][0], aoff, aoff + 8);
#pragma unroll
            for (int j = 0; j < 4; ++j) {
                acc[i][j] = __builtin_amdgcn_wmma_f32_16x16x32_bf16(
                    false, fah, false, fbh[j], (short)0, acc[i][j], false, false);
                acc[i][j] = __builtin_amdgcn_wmma_f32_16x16x32_bf16(
                    false, fah, false, fbl[j], (short)0, acc[i][j], false, false);
                acc[i][j] = __builtin_amdgcn_wmma_f32_16x16x32_bf16(
                    false, fal, false, fbh[j], (short)0, acc[i][j], false, false);
            }
        }
        __syncthreads();
    }
#endif

    // ---- epilogue: bias add + store ----
#pragma unroll
    for (int j = 0; j < 4; ++j) {
        const int col = n0 + wn * 64 + j * 16 + l16;
        const float bb = Bv[col];
#pragma unroll
        for (int i = 0; i < 4; ++i) {
            const int rowb = m0 + wm * 64 + i * 16 + hh * 8;
#pragma unroll
            for (int v = 0; v < 8; ++v)
                Y[(size_t)(rowb + v) * OUT_F + col] = acc[i][j][v] + bb;
        }
    }
}

// ================= fallback: fused convert-in-loop (round-1) ===============

#define FBM 128
#define FBN 128
#define FBK 32
#define FLDK 40

__global__ __launch_bounds__(256)
void gemm_fused_fallback(const float* __restrict__ X,
                         const float* __restrict__ W,
                         const unsigned char* __restrict__ Mk,
                         const float* __restrict__ Bv,
                         float* __restrict__ Y) {
    __shared__ unsigned short Ahi[FBM][FLDK];
    __shared__ unsigned short Alo[FBM][FLDK];
    __shared__ unsigned short Bhi[FBN][FLDK];
    __shared__ unsigned short Blo[FBN][FLDK];

    const int tid = threadIdx.x;
    const int m0  = blockIdx.y * FBM;
    const int n0  = blockIdx.x * FBN;
    const int lr  = tid >> 1;
    const int lk  = (tid & 1) << 4;
    const int lane = tid & 31;
    const int wv   = tid >> 5;
    const int wm   = wv >> 2;
    const int wn   = wv & 3;
    const int l16  = lane & 15;
    const int hh   = (lane >> 4) & 1;
    const int aoff = hh * 16;
    const int boff = hh * 16;

    v8f acc[4][2];
#pragma unroll
    for (int i = 0; i < 4; ++i)
#pragma unroll
        for (int j = 0; j < 2; ++j)
#pragma unroll
            for (int v = 0; v < 8; ++v) acc[i][j][v] = 0.0f;

    for (int k0 = 0; k0 < IN_F; k0 += FBK) {
        {
            const float* xp = X + (size_t)(m0 + lr) * IN_F + k0 + lk;
            float xr[16];
#pragma unroll
            for (int u = 0; u < 4; ++u) *(float4*)&xr[u * 4] = ((const float4*)xp)[u];
            unsigned* ah = (unsigned*)&Ahi[lr][lk];
            unsigned* al = (unsigned*)&Alo[lr][lk];
#pragma unroll
            for (int u = 0; u < 8; ++u) {
                unsigned hp, lp;
                split_pack(xr[2 * u], xr[2 * u + 1], hp, lp);
                ah[u] = hp; al[u] = lp;
            }
        }
        {
            const float* wp = W + (size_t)(n0 + lr) * IN_F + k0 + lk;
            const unsigned char* mp = Mk + (size_t)(n0 + lr) * IN_F + k0 + lk;
            float wr[16];
#pragma unroll
            for (int u = 0; u < 4; ++u) *(float4*)&wr[u * 4] = ((const float4*)wp)[u];
            uint4 mv = *(const uint4*)mp;
            unsigned mw[4] = {mv.x, mv.y, mv.z, mv.w};
#pragma unroll
            for (int u = 0; u < 16; ++u)
                if (((mw[u >> 2] >> ((u & 3) * 8)) & 0xFFu) == 0u) wr[u] = 0.0f;
            unsigned* bh = (unsigned*)&Bhi[lr][lk];
            unsigned* bl = (unsigned*)&Blo[lr][lk];
#pragma unroll
            for (int u = 0; u < 8; ++u) {
                unsigned hp, lp;
                split_pack(wr[2 * u], wr[2 * u + 1], hp, lp);
                bh[u] = hp; bl[u] = lp;
            }
        }
        __syncthreads();

        v16bf fbh[2], fbl[2];
#pragma unroll
        for (int j = 0; j < 2; ++j) {
            const int nc = wn * 32 + j * 16 + l16;
            fbh[j] = ld_frag(&Bhi[nc][0], boff, boff + 8);
            fbl[j] = ld_frag(&Blo[nc][0], boff, boff + 8);
        }
#pragma unroll
        for (int i = 0; i < 4; ++i) {
            const int mr = wm * 64 + i * 16 + l16;
            v16bf fah = ld_frag(&Ahi[mr][0], aoff, aoff + 8);
            v16bf fal = ld_frag(&Alo[mr][0], aoff, aoff + 8);
#pragma unroll
            for (int j = 0; j < 2; ++j) {
                acc[i][j] = __builtin_amdgcn_wmma_f32_16x16x32_bf16(
                    false, fah, false, fbh[j], (short)0, acc[i][j], false, false);
                acc[i][j] = __builtin_amdgcn_wmma_f32_16x16x32_bf16(
                    false, fah, false, fbl[j], (short)0, acc[i][j], false, false);
                acc[i][j] = __builtin_amdgcn_wmma_f32_16x16x32_bf16(
                    false, fal, false, fbh[j], (short)0, acc[i][j], false, false);
            }
        }
        __syncthreads();
    }

#pragma unroll
    for (int j = 0; j < 2; ++j) {
        const int col = n0 + wn * 32 + j * 16 + l16;
        const float bb = Bv[col];
#pragma unroll
        for (int i = 0; i < 4; ++i) {
            const int rowb = m0 + wm * 64 + i * 16 + hh * 8;
#pragma unroll
            for (int v = 0; v < 8; ++v)
                Y[(size_t)(rowb + v) * OUT_F + col] = acc[i][j][v] + bb;
        }
    }
}

// =========================== host launcher =================================

extern "C" void kernel_launch(void* const* d_in, const int* in_sizes, int n_in,
                              void* d_out, int out_size, void* d_ws, size_t ws_size,
                              hipStream_t stream) {
    (void)in_sizes; (void)n_in; (void)out_size;
    const float*         X  = (const float*)d_in[0];
    const float*         W  = (const float*)d_in[1];
    const unsigned char* Mk = (const unsigned char*)d_in[2];  // JAX bool = 1 byte
    const float*         Bv = (const float*)d_in[3];
    float*               Y  = (float*)d_out;

    const size_t xElems = (size_t)TOK * IN_F;     // 33.5M
    const size_t wElems = (size_t)OUT_F * IN_F;   // 16.8M
    const size_t need   = (2 * xElems + 2 * wElems) * sizeof(unsigned short); // 192 MiB

    if (ws_size >= need && d_ws != nullptr) {
        char* base = (char*)d_ws;
        unsigned short* Xh = (unsigned short*)(base);
        unsigned short* Xl = (unsigned short*)(base + xElems * 2);
        unsigned short* Wh = (unsigned short*)(base + 2 * xElems * 2);
        unsigned short* Wl = (unsigned short*)(base + 2 * xElems * 2 + wElems * 2);

        prep_x<<<dim3((unsigned)(xElems / (256 * 8))), dim3(256), 0, stream>>>(X, Xh, Xl);
        prep_w<<<dim3((unsigned)(wElems / (256 * 8))), dim3(256), 0, stream>>>(W, Mk, Wh, Wl);

        dim3 grid(OUT_F / BN, TOK / BM);   // 16 x 32 blocks
        gemm_bf16x3<<<grid, dim3(512), 0, stream>>>(Xh, Xl, Wh, Wl, Bv, Y);
    } else {
        dim3 grid(OUT_F / FBN, TOK / FBM); // 32 x 64 blocks
        gemm_fused_fallback<<<grid, dim3(256), 0, stream>>>(X, W, Mk, Bv, Y);
    }
}
